// AminoSubBlock_16003048145214
// MI455X (gfx1250) — matrix operand order repaired and checked by reading, not compiled, not run on hardware
//
#include <hip/hip_runtime.h>
#include <math.h>

// ---------------------------------------------------------------------------
// CDNA5 (gfx1250) implementation of the equivariant amino block.
// All channel contractions run through v_wmma_f32_16x16x32_bf16 (wave32),
// with LDS-staged bf16 operand tiles (ds_read_b128 fragment loads).
// Each wave computes a 16x32 output tile (2 accumulators, 2 wmma per K-step).
// ---------------------------------------------------------------------------

typedef __attribute__((ext_vector_type(16))) __bf16 bf16x16;
typedef __attribute__((ext_vector_type(8)))  __bf16 bf16x8;
typedef __attribute__((ext_vector_type(8)))  float  floatx8;

#define DIM_A 128
#define DIM_V 64
#define DIM_D 32
#define NATOM 14
#define NB    8
#define NM    512
#define BM    4096   // NB*NM   (rows of every GEMM in this net)
#define NN    7168   // NM*NATOM
#define LRELU 0.1f

#define LDS_STRIDE 72   // bf16 elements per staged row (36 dwords; bank-spread)

__device__ __forceinline__ unsigned short bf16bits(float f) {
  union { float f; unsigned u; } c; c.f = f;
  unsigned u = c.u;
  return (unsigned short)((u + 0x7FFFu + ((u >> 16) & 1u)) >> 16);
}
__device__ __forceinline__ unsigned pack2(float lo, float hi) {
  return (unsigned)bf16bits(lo) | ((unsigned)bf16bits(hi) << 16);
}

__device__ __forceinline__ bf16x16 frag16(const __bf16* p0, const __bf16* p1) {
  bf16x8 lo = *(const bf16x8*)p0;
  bf16x8 hi = *(const bf16x8*)p1;
  return __builtin_shufflevector(lo, hi,
      0, 1, 2, 3, 4, 5, 6, 7, 8, 9, 10, 11, 12, 13, 14, 15);
}

// Stage one KC-wide chunk (A: 128 rows, B: 32 cols) into LDS as packed bf16.
// Rows masked invalid (conv boundaries) were zero-filled once before the
// K-loop and are simply skipped here.
template <int KC>
__device__ __forceinline__ void stage_chunk(
    int k0, int K,
    const float* __restrict__ aBase, long xK, bool aValid,
    int aRow, int aHalfSel, unsigned* As32,
    const float* __restrict__ bBase, long wK,
    int bCol, int bSub, unsigned* Bs32)
{
  // ---- A: 128 rows x KC ----
  if (aValid) {
    constexpr int cnt = KC / 2;               // bf16 per thread
    const int ks = aHalfSel * cnt;
    unsigned* dst = As32 + ((aRow * LDS_STRIDE + ks) >> 1);
    if (xK == 1) {
      const float* xp = aBase + (k0 + ks);
#pragma unroll
      for (int i = 0; i < cnt; i += 4) {
        float4 v = *(const float4*)(xp + i);
        dst[(i >> 1)]     = pack2(v.x, v.y);
        dst[(i >> 1) + 1] = pack2(v.z, v.w);
      }
      if (k0 + KC < K) __builtin_prefetch(xp + KC, 0, 1);
    } else {
      const float* xp = aBase + (long)(k0 + ks) * xK;
#pragma unroll
      for (int i = 0; i < cnt; i += 2) {
        float a0 = xp[0];
        float a1 = xp[xK];
        xp += 2 * xK;
        dst[i >> 1] = pack2(a0, a1);
      }
    }
  }
  // ---- B: 32 cols x KC ----
  {
    constexpr int cntB = KC / 8;              // bf16 per thread
    const int kb0 = bSub * cntB;
    unsigned* dstB = Bs32 + ((bCol * LDS_STRIDE + kb0) >> 1);
    if (wK == 1) {
      const float* wp = bBase + (k0 + kb0);
#pragma unroll
      for (int i = 0; i < cntB; i += 4) {
        float4 v = *(const float4*)(wp + i);
        dstB[(i >> 1)]     = pack2(v.x, v.y);
        dstB[(i >> 1) + 1] = pack2(v.z, v.w);
      }
    } else {
      const float* wp = bBase + (long)(k0 + kb0) * wK;
#pragma unroll
      for (int i = 0; i < cntB; i += 2) {
        float b0 = wp[0];
        float b1 = wp[wK];
        wp += 2 * wK;
        dstB[i >> 1] = pack2(b0, b1);
      }
    }
  }
}

// ---------------------------------------------------------------------------
// Strided WMMA GEMM:  Out[r,n,z] = alpha * sum_k X[r,k,z] * W[n,k,z]
//                                + (init / accumulate) + bias[n]
// rows fixed at 4096.  Block = 256 threads = 8 waves covering 128 consecutive
// rows x 32 output columns.  Branch-free 64-wide main K loop (4 wmma per
// iteration on two accumulators) + straight-line 32-wide tail when K%64==32.
// Conv taps via (convShift, mMod) row masking resolved at staging time;
// transComp permutes the rank-2 component index on store.
// ---------------------------------------------------------------------------
__global__ __launch_bounds__(256) void wmma_gemm_kernel(
    const float* __restrict__ X, long xRow, long xK, long xZ,
    const float* __restrict__ W, long wN, long wK, long wZ,
    float* __restrict__ Out, long oRow, long oN, long oZ,
    const float* __restrict__ Init, long iRow, long iN, long iZ,
    const float* __restrict__ bias,
    int K, int Ncols,
    float alpha, float beta, int initMode,   // 0: none, 1: beta*Init, 2: += Out
    int convShift, int mMod, int transComp)
{
  __shared__ __align__(16) __bf16 As[128 * LDS_STRIDE];
  __shared__ __align__(16) __bf16 Bs[32 * LDS_STRIDE];
  unsigned* As32 = (unsigned*)As;
  unsigned* Bs32 = (unsigned*)Bs;

  const int tid  = threadIdx.x;
  const int lane = tid & 31;
  const int wave = tid >> 5;
  const int half = lane >> 4;
  const int l15  = lane & 15;

  const int tn   = blockIdx.x >> 5;          // 32-column group (shared by block)
  const int row0 = (blockIdx.x & 31) * 128;  // first of 128 staged rows
  const int z    = blockIdx.z;
  const int zo   = transComp ? ((z % 3) * 3 + (z / 3)) : z;

  // --- A staging map: thread -> (row, K-half) ---
  const int aRow = tid >> 1;                 // 0..127
  const int aHalfSel = tid & 1;
  int aGRow = row0 + aRow;
  int aSrcRow = aGRow + convShift;
  bool aValid = true;
  if (mMod > 0) {
    int m = (aGRow % mMod) + convShift;
    aValid = (m >= 0) && (m < mMod);
    if (!aValid) aSrcRow = aGRow;
  }
  const float* aBase = X + (long)aSrcRow * xRow + (long)z * xZ;

  // Invalid conv rows contribute zero: fill their LDS row once (stays zero
  // for the whole kernel; only this thread pair ever writes that row).
  if (!aValid) {
    unsigned* dst = As32 + ((aRow * LDS_STRIDE + aHalfSel * 32) >> 1);
#pragma unroll
    for (int i = 0; i < 16; ++i) dst[i] = 0u;
  }

  // --- B staging map: thread -> (col, K-eighth) ---
  const int bCol = tid >> 3;                 // 0..31
  const int bSub = tid & 7;
  const float* bBase = W + (long)(tn * 32 + bCol) * wN + (long)z * wZ;

  floatx8 acc0 = {0.f, 0.f, 0.f, 0.f, 0.f, 0.f, 0.f, 0.f};
  floatx8 acc1 = {0.f, 0.f, 0.f, 0.f, 0.f, 0.f, 0.f, 0.f};

  // A 16x32 bf16 layout: half 0 -> K 0..15 ; half 1 -> K 16..31
  // B 32x16 bf16 layout: half 0 -> K 0..15 ; half 1 -> K 16..31
  const __bf16* a0p = As + (wave * 16 + l15) * LDS_STRIDE + 16 * half;
  const __bf16* b0p = Bs + l15 * LDS_STRIDE + 16 * half;
  const __bf16* b1p = Bs + (16 + l15) * LDS_STRIDE + 16 * half;

  const int K64 = K & ~63;
  int k0 = 0;
  for (; k0 < K64; k0 += 64) {
    stage_chunk<64>(k0, K, aBase, xK, aValid, aRow, aHalfSel, As32,
                    bBase, wK, bCol, bSub, Bs32);
    __syncthreads();
    bf16x16 a0  = frag16(a0p,      a0p + 8);
    bf16x16 a1  = frag16(a0p + 32, a0p + 40);
    bf16x16 b00 = frag16(b0p,      b0p + 8);
    bf16x16 b01 = frag16(b1p,      b1p + 8);
    bf16x16 b10 = frag16(b0p + 32, b0p + 40);
    bf16x16 b11 = frag16(b1p + 32, b1p + 40);
    acc0 = __builtin_amdgcn_wmma_f32_16x16x32_bf16(false, a0, false, b00, (short)0, acc0, false, false);
    acc1 = __builtin_amdgcn_wmma_f32_16x16x32_bf16(false, a0, false, b01, (short)0, acc1, false, false);
    acc0 = __builtin_amdgcn_wmma_f32_16x16x32_bf16(false, a1, false, b10, (short)0, acc0, false, false);
    acc1 = __builtin_amdgcn_wmma_f32_16x16x32_bf16(false, a1, false, b11, (short)0, acc1, false, false);
    __syncthreads();
  }
  if (K & 32) {   // single straight-line 32-wide tail (only K==32 cases here)
    stage_chunk<32>(k0, K, aBase, xK, aValid, aRow, aHalfSel, As32,
                    bBase, wK, bCol, bSub, Bs32);
    __syncthreads();
    bf16x16 a0  = frag16(a0p, a0p + 8);
    bf16x16 b00 = frag16(b0p, b0p + 8);
    bf16x16 b01 = frag16(b1p, b1p + 8);
    acc0 = __builtin_amdgcn_wmma_f32_16x16x32_bf16(false, a0, false, b00, (short)0, acc0, false, false);
    acc1 = __builtin_amdgcn_wmma_f32_16x16x32_bf16(false, a0, false, b01, (short)0, acc1, false, false);
  }

  // ---- epilogue: C/D layout VGPR rr -> row rr + 8*half, col = lane&15 ----
  const int ncol0 = tn * 32 + l15;
  const int ncol1 = ncol0 + 16;
#pragma unroll
  for (int rr = 0; rr < 8; ++rr) {
    int mrow = row0 + wave * 16 + rr + 8 * half;
    long rbase = (long)mrow * oRow + (long)zo * oZ;
    long o0 = rbase + (long)ncol0 * oN;
    long o1 = rbase + (long)ncol1 * oN;
    float v0 = alpha * acc0[rr];
    float v1 = alpha * acc1[rr];
    if (initMode == 1) {
      long ibase = (long)mrow * iRow + (long)zo * iZ;
      v0 += beta * Init[ibase + (long)ncol0 * iN];
      v1 += beta * Init[ibase + (long)ncol1 * iN];
    } else if (initMode == 2) {
      v0 += Out[o0];
      v1 += Out[o1];
    }
    if (bias) { v0 += bias[ncol0]; v1 += bias[ncol1]; }
    Out[o0] = v0;
    Out[o1] = v1;
  }
}

// ---------------------------------------------------------------------------
// Elementwise / small kernels
// ---------------------------------------------------------------------------
__global__ void k_leaky(float* x, long n) {
  long i = (long)blockIdx.x * blockDim.x + threadIdx.x;
  if (i < n) { float v = x[i]; x[i] = v >= 0.f ? v : LRELU * v; }
}

__global__ void k_add(float* dst, const float* src, long n) {
  long i = (long)blockIdx.x * blockDim.x + threadIdx.x;
  if (i < n) dst[i] += src[i];
}

// per-amino mean positions: pa[bm*3+x]
__global__ void k_pamean(const float* __restrict__ p0, const float* __restrict__ p1,
                         float* pa0, float* pa1) {
  int idx = blockIdx.x * blockDim.x + threadIdx.x;
  if (idx >= BM * 3) return;
  int x = idx % 3; int bm = idx / 3;
  int b = bm >> 9, m = bm & 511;
  long base = ((long)b * NN + (long)m * NATOM) * 3 + x;
  float s0 = 0.f, s1 = 0.f;
  for (int a = 0; a < NATOM; ++a) { s0 += p0[base + a * 3]; s1 += p1[base + a * 3]; }
  pa0[idx] = s0 * (1.f / NATOM);
  pa1[idx] = s1 * (1.f / NATOM);
}

// adds dv0 + dv1 + pe + nde into Yv;   grid = BM blocks, 64 threads (o)
__global__ void k_encode(const float* __restrict__ p0, const float* __restrict__ p1,
                         const float* __restrict__ pa0, const float* __restrict__ pa1,
                         const float* __restrict__ enc0, const float* __restrict__ enc1,
                         const float* __restrict__ posW,
                         const float* __restrict__ ndl, const float* __restrict__ ndr,
                         float* __restrict__ Yv) {
  int bm = blockIdx.x;
  int o  = threadIdx.x;
  int b = bm >> 9, m = bm & 511;
  long nbase = ((long)b * NN + (long)m * NATOM) * 3;
  float add[3];
  for (int x = 0; x < 3; ++x) {
    float c0 = pa0[bm * 3 + x], c1 = pa1[bm * 3 + x];
    float s0 = 0.f, s1 = 0.f;
    for (int a = 0; a < NATOM; ++a) {
      s0 += enc0[o * NATOM + a] * (p0[nbase + a * 3 + x] - c0);
      s1 += enc1[o * NATOM + a] * (p1[nbase + a * 3 + x] - c1);
    }
    float pe = posW[o] * 0.1f * (c1 - c0);
    float dl0 = 0.f, dl1 = 0.f, dr0 = 0.f, dr1 = 0.f;
    if (m < NM - 1) { dl0 = pa0[(bm + 1) * 3 + x] - c0; dl1 = pa1[(bm + 1) * 3 + x] - c1; }
    if (m > 0)      { dr0 = c0 - pa0[(bm - 1) * 3 + x]; dr1 = c1 - pa1[(bm - 1) * 3 + x]; }
    float nde = ndl[o * 2 + 0] * dl0 + ndl[o * 2 + 1] * dl1
              + ndr[o * 2 + 0] * dr0 + ndr[o * 2 + 1] * dr1;
    add[x] = s0 + s1 + pe + nde;
  }
  for (int x = 0; x < 3; ++x) Yv[(long)bm * 192 + o * 3 + x] += add[x];
}

// SO(3) tensor products; thread per (bm, c), c in [0,64)
__global__ void k_bee(int op,
                      const float* __restrict__ l0, const float* __restrict__ r0,
                      const float* __restrict__ l1, const float* __restrict__ r1,
                      const float* __restrict__ l2, const float* __restrict__ r2,
                      float* __restrict__ out) {
  long idx = (long)blockIdx.x * blockDim.x + threadIdx.x;
  if (idx >= (long)BM * 64) return;
  const float* L1 = l1 + idx * 3; const float* R1 = r1 + idx * 3;
  const float* L2 = l2 + idx * 9; const float* R2 = r2 + idx * 9;
  float s0 = l0[idx], t0 = r0[idx];
  switch (op) {
    case 0: out[idx] = s0 * t0; break;
    case 1: { float s = 0; for (int x = 0; x < 3; ++x) s += L1[x] * R1[x]; out[idx] = s; } break;
    case 2: { float s = 0; for (int k = 0; k < 9; ++k) s += L2[k] * R2[k]; out[idx] = s; } break;
    case 3: for (int x = 0; x < 3; ++x) out[idx * 3 + x] = s0 * R1[x]; break;
    case 4: for (int x = 0; x < 3; ++x) out[idx * 3 + x] = L1[x] * t0; break;
    case 5: for (int j = 0; j < 3; ++j) { float s = 0; for (int i = 0; i < 3; ++i) s += L1[i] * R2[i * 3 + j]; out[idx * 3 + j] = s; } break;
    case 6: for (int i = 0; i < 3; ++i) { float s = 0; for (int j = 0; j < 3; ++j) s += L2[i * 3 + j] * R1[j]; out[idx * 3 + i] = s; } break;
    case 7: for (int k = 0; k < 9; ++k) out[idx * 9 + k] = s0 * R2[k]; break;
    case 8: for (int k = 0; k < 9; ++k) out[idx * 9 + k] = L2[k] * t0; break;
    case 9: for (int i = 0; i < 3; ++i) for (int j = 0; j < 3; ++j) out[idx * 9 + i * 3 + j] = L1[i] * R1[j]; break;
    case 10: for (int i = 0; i < 3; ++i) for (int j = 0; j < 3; ++j) { float s = 0; for (int k = 0; k < 3; ++k) s += L2[i * 3 + k] * R2[k * 3 + j]; out[idx * 9 + i * 3 + j] = s; } break;
    case 11:
      out[idx * 3 + 0] = L1[1] * R1[2] - L1[2] * R1[1];
      out[idx * 3 + 1] = L1[2] * R1[0] - L1[0] * R1[2];
      out[idx * 3 + 2] = L1[0] * R1[1] - L1[1] * R1[0];
      break;
    case 12:
      for (int i = 0; i < 3; ++i) {
        const float* u = L2 + i * 3;
        out[idx * 9 + i * 3 + 0] = u[1] * R1[2] - u[2] * R1[1];
        out[idx * 9 + i * 3 + 1] = u[2] * R1[0] - u[0] * R1[2];
        out[idx * 9 + i * 3 + 2] = u[0] * R1[1] - u[1] * R1[0];
      }
      break;
  }
}

// group norm (groups=8) over (d/8 channels x ncomp comps); thread per (bm, g)
__global__ void k_gnorm(float* __restrict__ x, const float* __restrict__ gamma,
                        int d, int ncomp, int subMean) {
  int idx = blockIdx.x * blockDim.x + threadIdx.x;
  const int groups = 8;
  if (idx >= BM * groups) return;
  int g = idx % groups; long bm = idx / groups;
  int cpg = d / groups;
  float* base = x + (bm * (long)d + g * cpg) * ncomp;
  int n = cpg * ncomp;
  float mean = 0.f;
  if (subMean) { float s = 0.f; for (int i = 0; i < n; ++i) s += base[i]; mean = s / n; }
  float v = 0.f;
  for (int i = 0; i < n; ++i) { float z = base[i] - mean; v += z * z; }
  v /= n;
  float inv = 1.f / sqrtf(v + 1e-5f);
  for (int i = 0; i < n; ++i) {
    int c = g * cpg + i / ncomp;
    base[i] = (base[i] - mean) * inv * gamma[c];
  }
}

// time embedding: emb[b*d+o] = sum_j sin/cos(t*freq_j) * We[o,j] + be[o]
__global__ void k_tembed(const float* __restrict__ t, const float* __restrict__ We,
                         const float* __restrict__ be, int d, float* __restrict__ emb) {
  int idx = blockIdx.x * blockDim.x + threadIdx.x;
  if (idx >= NB * d) return;
  int o = idx % d, b = idx / d;
  float tv = t[b], s = 0.f;
  for (int j = 0; j < 32; ++j) {
    float ph = tv * expf(-(float)j * (1.f / 31.f));
    s += sinf(ph) * We[o * 64 + j] + cosf(ph) * We[o * 64 + 32 + j];
  }
  emb[idx] = s + be[o];
}

__global__ void k_mod(float* __restrict__ xi, const float* __restrict__ emb,
                      int d, int ncomp, long n) {
  long idx = (long)blockIdx.x * blockDim.x + threadIdx.x;
  if (idx >= n) return;
  long row = idx / ((long)d * ncomp);
  int o = (int)((idx / ncomp) % d);
  int b = (int)(row >> 9);
  xi[idx] *= emb[b * d + o];
}

// gated equivariant sigmoid on 3-vectors; thread per (bm, c)
__global__ void k_tsig1(float* __restrict__ v, long npairs) {
  long idx = (long)blockIdx.x * blockDim.x + threadIdx.x;
  if (idx >= npairs) return;
  float* p = v + idx * 3;
  float nn = sqrtf(p[0] * p[0] + p[1] * p[1] + p[2] * p[2]);
  float s = 1.f / (1.f + expf(-nn));
  p[0] *= s; p[1] *= s; p[2] *= s;
}

// ---------------------------------------------------------------------------
// Host orchestration
// ---------------------------------------------------------------------------
static inline long cdiv(long a, long b) { return (a + b - 1) / b; }

struct GemmCtx { hipStream_t stream; };

static void gemm(const GemmCtx& c,
                 const float* X, long xRow, long xK, long xZ,
                 const float* W, long wN, long wK, long wZ,
                 float* O, long oRow, long oN, long oZ,
                 const float* Init, long iRow, long iN, long iZ,
                 const float* bias, int rows, int K, int N, int nz,
                 float alpha, float beta, int initMode,
                 int convShift = 0, int mMod = 0, int transComp = 0) {
  (void)rows;                       // all GEMMs are 4096 rows (32 row-blocks)
  dim3 grid((unsigned)(32 * (N / 32)), 1, (unsigned)nz);
  wmma_gemm_kernel<<<grid, 256, 0, c.stream>>>(
      X, xRow, xK, xZ, W, wN, wK, wZ, O, oRow, oN, oZ,
      Init, iRow, iN, iZ, bias, K, N,
      alpha, beta, initMode, convShift, mMod, transComp);
}

extern "C" void kernel_launch(void* const* d_in, const int* in_sizes, int n_in,
                              void* d_out_v, int out_size, void* d_ws, size_t ws_size,
                              hipStream_t stream) {
  (void)in_sizes; (void)out_size; (void)ws_size;
  GemmCtx C{stream};

  const float* t_in = (const float*)d_in[0];
  const float* x_a  = (const float*)d_in[1];
  const float* x_v  = (const float*)d_in[2];
  const float* x_d  = (const float*)d_in[3];
  const float* yain = (const float*)d_in[4];
  const float* yvin = (const float*)d_in[5];
  const float* ydin = (const float*)d_in[6];
  const float* pos0 = (const float*)d_in[7];
  const float* pos1 = (const float*)d_in[8];
  // d_in[9] = atoms_per_amino (compile-time 14)

  // params in _make_params insertion order, starting at d_in[10]
  const float* P[64];
  for (int i = 10; i < n_in && i < 74; ++i) P[i - 10] = (const float*)d_in[i];
  enum {
    readin_a = 0, readin_v, readin_d, enc0, enc1, pos_W, ndisp_l, ndisp_r,
    conv_a1, conv_a2, conv_a3, conv_v1, conv_v2, conv_d1, conv_d2,
    L0, R0, L1, R1, L2, R2,
    Y_000, Y_110, Y_220, Y_011, Y_101, Y_121, Y_211, Y_111,
    Y_022, Y_202, Y_112, Y_222, Y_212,
    gn_a_g, gn_v_g, gn_d_g,
    tlm_a_We, tlm_a_be, tlm_a_Wi, tlm_a_Wo,
    tlm_v_We, tlm_v_be, tlm_v_Wi, tlm_v_Wo,
    tlm_d_We, tlm_d_be, tlm_d_Wi, tlm_d_Wo,
    mlp_a_dir, mlp_a_w1, mlp_a_b1, mlp_a_w2, mlp_a_b2, mlp_a_w3, mlp_a_b3,
    mlp_v_dir, mlp_v_w1, mlp_v_w2, mlp_v_w3,
    lin_d_trans, readout_a, readout_v, readout_d
  };

  float* out = (float*)d_out_v;

  // workspace carve-up (floats)
  float* w = (float*)d_ws;
  size_t off = 0;
  auto alloc = [&](size_t n) { float* p = w + off; off += n; return p; };
  float* Ya  = alloc((size_t)BM * 128);
  float* Yv  = alloc((size_t)BM * 192);
  float* Yd  = alloc((size_t)BM * 288);
  float* pa  = alloc((size_t)BM * 128);
  float* pv  = alloc((size_t)BM * 192);
  float* pd  = alloc((size_t)BM * 288);
  float* pa2 = alloc((size_t)BM * 128);
  float* pv2 = alloc((size_t)BM * 192);
  float* pd2 = alloc((size_t)BM * 288);
  float* l0b = alloc((size_t)BM * 64);
  float* r0b = alloc((size_t)BM * 64);
  float* l1b = alloc((size_t)BM * 192);
  float* r1b = alloc((size_t)BM * 192);
  float* l2b = alloc((size_t)BM * 576);
  float* r2b = alloc((size_t)BM * 576);
  float* sc1 = alloc((size_t)BM * 576);
  float* sc2 = alloc((size_t)BM * 576);
  float* pa0b = alloc((size_t)BM * 3);
  float* pa1b = alloc((size_t)BM * 3);
  float* embA = alloc((size_t)NB * 128);
  float* embV = alloc((size_t)NB * 64);
  float* embD = alloc((size_t)NB * 32);

  auto ew = [&](long n) { return dim3((unsigned)cdiv(n, 256)); };

  // ---- read-in: Y = y + 0.2 * atom_to_amino(W, x) --------------------------
  gemm(C, x_a, 1792, 1, 0, P[readin_a], 1792, 1, 0, Ya, 128, 1, 0,
       yain, 128, 1, 0, nullptr, BM, 1792, 128, 1, 0.2f, 1.f, 1);
  gemm(C, x_v, 2688, 3, 1, P[readin_v], 896, 1, 0, Yv, 192, 3, 1,
       yvin, 192, 3, 1, nullptr, BM, 896, 64, 3, 0.2f, 1.f, 1);
  gemm(C, x_d, 4032, 9, 1, P[readin_d], 448, 1, 0, Yd, 288, 9, 1,
       ydin, 288, 9, 1, nullptr, BM, 448, 32, 9, 0.2f, 1.f, 1);

  // ---- positional encoders -> Yv ------------------------------------------
  k_pamean<<<ew(BM * 3), 256, 0, stream>>>(pos0, pos1, pa0b, pa1b);
  k_encode<<<BM, 64, 0, stream>>>(pos0, pos1, pa0b, pa1b, P[enc0], P[enc1],
                                  P[pos_W], P[ndisp_l], P[ndisp_r], Yv);

  // ---- chain convolutions (7-tap, SAME) -----------------------------------
  for (int tp = 0; tp < 7; ++tp)
    gemm(C, Ya, 128, 1, 0, P[conv_a1] + tp, 896, 7, 0, sc1, 128, 1, 0,
         nullptr, 0, 0, 0, nullptr, BM, 128, 128, 1, 1.f, 0.f, tp ? 2 : 0, tp - 3, NM);
  k_leaky<<<ew((long)BM * 128), 256, 0, stream>>>(sc1, (long)BM * 128);
  for (int tp = 0; tp < 7; ++tp)
    gemm(C, sc1, 128, 1, 0, P[conv_a2] + tp, 896, 7, 0, sc2, 128, 1, 0,
         nullptr, 0, 0, 0, nullptr, BM, 128, 128, 1, 1.f, 0.f, tp ? 2 : 0, tp - 3, NM);
  k_leaky<<<ew((long)BM * 128), 256, 0, stream>>>(sc2, (long)BM * 128);
  for (int tp = 0; tp < 7; ++tp)   // accumulate straight onto Ya (residual)
    gemm(C, sc2, 128, 1, 0, P[conv_a3] + tp, 896, 7, 0, Ya, 128, 1, 0,
         nullptr, 0, 0, 0, nullptr, BM, 128, 128, 1, 1.f, 0.f, 2, tp - 3, NM);

  for (int tp = 0; tp < 7; ++tp)
    gemm(C, Yv, 192, 3, 1, P[conv_v1] + tp, 448, 7, 0, sc1, 192, 3, 1,
         nullptr, 0, 0, 0, nullptr, BM, 64, 64, 3, 1.f, 0.f, tp ? 2 : 0, tp - 3, NM);
  for (int tp = 0; tp < 7; ++tp)
    gemm(C, sc1, 192, 3, 1, P[conv_v2] + tp, 448, 7, 0, Yv, 192, 3, 1,
         nullptr, 0, 0, 0, nullptr, BM, 64, 64, 3, 1.f, 0.f, 2, tp - 3, NM);

  for (int tp = 0; tp < 7; ++tp)
    gemm(C, Yd, 288, 9, 1, P[conv_d1] + tp, 224, 7, 0, sc1, 288, 9, 1,
         nullptr, 0, 0, 0, nullptr, BM, 32, 32, 9, 1.f, 0.f, tp ? 2 : 0, tp - 3, NM);
  for (int tp = 0; tp < 7; ++tp)
    gemm(C, sc1, 288, 9, 1, P[conv_d2] + tp, 224, 7, 0, Yd, 288, 9, 1,
         nullptr, 0, 0, 0, nullptr, BM, 32, 32, 9, 1.f, 0.f, 2, tp - 3, NM);

  // ---- L/R channel projections (ch=64) ------------------------------------
  gemm(C, Ya, 128, 1, 0, P[L0], 128, 1, 0, l0b, 64, 1, 0, nullptr, 0, 0, 0,
       nullptr, BM, 128, 64, 1, 1.f, 0.f, 0);
  gemm(C, Ya, 128, 1, 0, P[R0], 128, 1, 0, r0b, 64, 1, 0, nullptr, 0, 0, 0,
       nullptr, BM, 128, 64, 1, 1.f, 0.f, 0);
  gemm(C, Yv, 192, 3, 1, P[L1], 64, 1, 0, l1b, 192, 3, 1, nullptr, 0, 0, 0,
       nullptr, BM, 64, 64, 3, 1.f, 0.f, 0);
  gemm(C, Yv, 192, 3, 1, P[R1], 64, 1, 0, r1b, 192, 3, 1, nullptr, 0, 0, 0,
       nullptr, BM, 64, 64, 3, 1.f, 0.f, 0);
  gemm(C, Yd, 288, 9, 1, P[L2], 32, 1, 0, l2b, 576, 9, 1, nullptr, 0, 0, 0,
       nullptr, BM, 32, 64, 9, 1.f, 0.f, 0);
  gemm(C, Yd, 288, 9, 1, P[R2], 32, 1, 0, r2b, 576, 9, 1, nullptr, 0, 0, 0,
       nullptr, BM, 32, 64, 9, 1.f, 0.f, 0);

  // ---- tensor products -> Y-projections into pa/pv/pd ---------------------
  long beeN = (long)BM * 64;
  struct BeeOp { int op; const float* Wp; float* dst; long oR, oN2, oZ; int dOut, nz, im; };
  const BeeOp ops[13] = {
    {0,  P[Y_000], pa, 128, 1, 0, 128, 1, 0}, {1, P[Y_110], pa, 128, 1, 0, 128, 1, 2},
    {2,  P[Y_220], pa, 128, 1, 0, 128, 1, 2},
    {3,  P[Y_011], pv, 192, 3, 1, 64, 3, 0}, {4, P[Y_101], pv, 192, 3, 1, 64, 3, 2},
    {5,  P[Y_121], pv, 192, 3, 1, 64, 3, 2}, {6, P[Y_211], pv, 192, 3, 1, 64, 3, 2},
    {11, P[Y_111], pv, 192, 3, 1, 64, 3, 2},
    {7,  P[Y_022], pd, 288, 9, 1, 32, 9, 0}, {8, P[Y_202], pd, 288, 9, 1, 32, 9, 2},
    {9,  P[Y_112], pd, 288, 9, 1, 32, 9, 2}, {10, P[Y_222], pd, 288, 9, 1, 32, 9, 2},
    {12, P[Y_212], pd, 288, 9, 1, 32, 9, 2},
  };
  for (int i = 0; i < 13; ++i) {
    const BeeOp& o = ops[i];
    k_bee<<<ew(beeN), 256, 0, stream>>>(o.op, l0b, r0b, l1b, r1b, l2b, r2b, sc1);
    long sR = (o.nz == 1) ? 64 : ((o.nz == 3) ? 192 : 576);
    gemm(C, sc1, sR, o.nz, 1, o.Wp, 64, 1, 0, o.dst, o.oR, o.oN2, o.oZ,
         nullptr, 0, 0, 0, nullptr, BM, 64, o.dOut, o.nz, 1.f, 0.f, o.im);
  }

  // ---- group norms ---------------------------------------------------------
  k_gnorm<<<ew(BM * 8), 256, 0, stream>>>(pa, P[gn_a_g], 128, 1, 1);
  k_gnorm<<<ew(BM * 8), 256, 0, stream>>>(pv, P[gn_v_g], 64, 3, 0);
  k_gnorm<<<ew(BM * 8), 256, 0, stream>>>(pd, P[gn_d_g], 32, 9, 0);

  // ---- time-modulated linears (tlm) ---------------------------------------
  k_tembed<<<ew(NB * 128), 256, 0, stream>>>(t_in, P[tlm_a_We], P[tlm_a_be], 128, embA);
  k_tembed<<<ew(NB * 64),  256, 0, stream>>>(t_in, P[tlm_v_We], P[tlm_v_be], 64, embV);
  k_tembed<<<ew(NB * 32),  256, 0, stream>>>(t_in, P[tlm_d_We], P[tlm_d_be], 32, embD);

  gemm(C, pa, 128, 1, 0, P[tlm_a_Wi], 128, 1, 0, sc1, 128, 1, 0, nullptr, 0, 0, 0,
       nullptr, BM, 128, 128, 1, 1.f, 0.f, 0);
  k_mod<<<ew((long)BM * 128), 256, 0, stream>>>(sc1, embA, 128, 1, (long)BM * 128);
  gemm(C, sc1, 128, 1, 0, P[tlm_a_Wo], 128, 1, 0, pa, 128, 1, 0, nullptr, 0, 0, 0,
       nullptr, BM, 128, 128, 1, 1.f, 0.f, 2);

  gemm(C, pv, 192, 3, 1, P[tlm_v_Wi], 64, 1, 0, sc1, 192, 3, 1, nullptr, 0, 0, 0,
       nullptr, BM, 64, 64, 3, 1.f, 0.f, 0);
  k_mod<<<ew((long)BM * 192), 256, 0, stream>>>(sc1, embV, 64, 3, (long)BM * 192);
  gemm(C, sc1, 192, 3, 1, P[tlm_v_Wo], 64, 1, 0, pv, 192, 3, 1, nullptr, 0, 0, 0,
       nullptr, BM, 64, 64, 3, 1.f, 0.f, 2);

  gemm(C, pd, 288, 9, 1, P[tlm_d_Wi], 32, 1, 0, sc1, 288, 9, 1, nullptr, 0, 0, 0,
       nullptr, BM, 32, 32, 9, 1.f, 0.f, 0);
  k_mod<<<ew((long)BM * 288), 256, 0, stream>>>(sc1, embD, 32, 9, (long)BM * 288);
  gemm(C, sc1, 288, 9, 1, P[tlm_d_Wo], 32, 1, 0, pd, 288, 9, 1, nullptr, 0, 0, 0,
       nullptr, BM, 32, 32, 9, 1.f, 0.f, 2);

  // ---- scalar MLP ----------------------------------------------------------
  gemm(C, pa, 128, 1, 0, P[mlp_a_w1], 128, 1, 0, sc1, 256, 1, 0, nullptr, 0, 0, 0,
       P[mlp_a_b1], BM, 128, 256, 1, 1.f, 0.f, 0);
  k_leaky<<<ew((long)BM * 256), 256, 0, stream>>>(sc1, (long)BM * 256);
  gemm(C, sc1, 256, 1, 0, P[mlp_a_w2], 256, 1, 0, sc2, 256, 1, 0, nullptr, 0, 0, 0,
       P[mlp_a_b2], BM, 256, 256, 1, 1.f, 0.f, 0);
  k_leaky<<<ew((long)BM * 256), 256, 0, stream>>>(sc2, (long)BM * 256);
  gemm(C, pa, 128, 1, 0, P[mlp_a_dir], 128, 1, 0, pa2, 128, 1, 0,
       pa, 128, 1, 0, nullptr, BM, 128, 128, 1, 1.f, 1.f, 1);
  gemm(C, sc2, 256, 1, 0, P[mlp_a_w3], 256, 1, 0, pa2, 128, 1, 0, nullptr, 0, 0, 0,
       P[mlp_a_b3], BM, 256, 128, 1, 1.f, 0.f, 2);

  // ---- vector MLP (gated sigmoids) ----------------------------------------
  gemm(C, pv, 192, 3, 1, P[mlp_v_w1], 64, 1, 0, sc1, 384, 3, 1, nullptr, 0, 0, 0,
       nullptr, BM, 64, 128, 3, 1.f, 0.f, 0);
  k_tsig1<<<ew((long)BM * 128), 256, 0, stream>>>(sc1, (long)BM * 128);
  gemm(C, sc1, 384, 3, 1, P[mlp_v_w2], 128, 1, 0, sc2, 384, 3, 1, nullptr, 0, 0, 0,
       nullptr, BM, 128, 128, 3, 1.f, 0.f, 0);
  k_tsig1<<<ew((long)BM * 128), 256, 0, stream>>>(sc2, (long)BM * 128);
  gemm(C, pv, 192, 3, 1, P[mlp_v_dir], 64, 1, 0, pv2, 192, 3, 1,
       pv, 192, 3, 1, nullptr, BM, 64, 64, 3, 1.f, 1.f, 1);
  gemm(C, sc2, 384, 3, 1, P[mlp_v_w3], 128, 1, 0, pv2, 192, 3, 1, nullptr, 0, 0, 0,
       nullptr, BM, 128, 64, 3, 1.f, 0.f, 2);

  // ---- rank-2 transpose mix: pd2 = pd + swap_ij(lin_d_trans(pd)) ----------
  gemm(C, pd, 288, 9, 1, P[lin_d_trans], 32, 1, 0, pd2, 288, 9, 1,
       pd, 288, 9, 1, nullptr, BM, 32, 32, 9, 1.f, 1.f, 1, 0, 0, /*transComp=*/1);

  // ---- residuals, y outputs -----------------------------------------------
  k_add<<<ew((long)BM * 128), 256, 0, stream>>>(Ya, pa2, (long)BM * 128);
  k_add<<<ew((long)BM * 192), 256, 0, stream>>>(Yv, pv2, (long)BM * 192);
  k_add<<<ew((long)BM * 288), 256, 0, stream>>>(Yd, pd2, (long)BM * 288);

  const long dxA = 0, dxV = 7340032L, dxD = 18350080L;
  const long oYa = 34865152L, oYv = 35389440L, oYd = 36175872L;
  hipMemcpyAsync(out + oYa, Ya, (size_t)BM * 128 * sizeof(float), hipMemcpyDeviceToDevice, stream);
  hipMemcpyAsync(out + oYv, Yv, (size_t)BM * 192 * sizeof(float), hipMemcpyDeviceToDevice, stream);
  hipMemcpyAsync(out + oYd, Yd, (size_t)BM * 288 * sizeof(float), hipMemcpyDeviceToDevice, stream);

  // ---- read-out: dx = 0.2 * amino_to_atom(readout, y) ---------------------
  for (int a = 0; a < NATOM; ++a) {
    gemm(C, Ya, 128, 1, 0, P[readout_a] + (long)a * 128 * 128, 128, 1, 0,
         out + dxA + (long)a * 128, 1792, 1, 0, nullptr, 0, 0, 0,
         nullptr, BM, 128, 128, 1, 0.2f, 0.f, 0);
    gemm(C, Yv, 192, 3, 1, P[readout_v] + (long)a * 64 * 64, 64, 1, 0,
         out + dxV + (long)a * 192, 2688, 3, 1, nullptr, 0, 0, 0,
         nullptr, BM, 64, 64, 3, 0.2f, 0.f, 0);
    gemm(C, Yd, 288, 9, 1, P[readout_d] + (long)a * 32 * 32, 32, 1, 0,
         out + dxD + (long)a * 288, 4032, 9, 1, nullptr, 0, 0, 0,
         nullptr, BM, 32, 32, 9, 0.2f, 0.f, 0);
  }
}
